// HybridQLSTM_65481071398269
// MI455X (gfx1250) — matrix operand order repaired and checked by reading, not compiled, not run on hardware
//
#include <hip/hip_runtime.h>
#include <hip/hip_fp16.h>

#define N_QUBITS 8
#define EMBED    512
#define SEQ      128
#define BATCH    256
#define DIN      (EMBED + N_QUBITS)   // 520
#define NCOLS    32                   // 4 gates * 8 outputs
#define KSTEPS   (EMBED / 32)         // 16 WMMA k-steps

typedef __attribute__((ext_vector_type(16))) _Float16 v16h;
typedef __attribute__((ext_vector_type(8)))  _Float16 v8h;
typedef __attribute__((ext_vector_type(8)))  float    v8f;

// Convert 8 f32 (two float4) -> v8h (one 16B LDS slot)
__device__ __forceinline__ v8h cvt8(float4 x0, float4 x1) {
    v8h r;
    r[0] = (_Float16)x0.x; r[1] = (_Float16)x0.y; r[2] = (_Float16)x0.z; r[3] = (_Float16)x0.w;
    r[4] = (_Float16)x1.x; r[5] = (_Float16)x1.y; r[6] = (_Float16)x1.z; r[7] = (_Float16)x1.w;
    return r;
}

__device__ __forceinline__ v16h cvt16(float4 x0, float4 x1, float4 x2, float4 x3) {
    v16h r;
    r[0] = (_Float16)x0.x; r[1] = (_Float16)x0.y; r[2] = (_Float16)x0.z; r[3] = (_Float16)x0.w;
    r[4] = (_Float16)x1.x; r[5] = (_Float16)x1.y; r[6] = (_Float16)x1.z; r[7] = (_Float16)x1.w;
    r[8] = (_Float16)x2.x; r[9] = (_Float16)x2.y; r[10]= (_Float16)x2.z; r[11]= (_Float16)x2.w;
    r[12]= (_Float16)x3.x; r[13]= (_Float16)x3.y; r[14]= (_Float16)x3.z; r[15]= (_Float16)x3.w;
    return r;
}

__device__ __forceinline__ v16h join16(v8h lo, v8h hi) {
    v16h r;
#pragma unroll
    for (int i = 0; i < 8; ++i) { r[i] = lo[i]; r[i + 8] = hi[i]; }
    return r;
}

// Kernel 1: Zx[(t*BATCH+b)][g*8+q] = emb[sent[t,b]] . W_g[q, 0:512]
// One wave per 16-row M tile; N=32 as two 16-col WMMA tiles.
// Full f16 B staged once per block in LDS in per-lane fragment layout.
__global__ void __launch_bounds__(256)
qlstm_gemm_x(const int*   __restrict__ sentence,
             const float* __restrict__ emb,
             const float* __restrict__ Wf,
             const float* __restrict__ Wi,
             const float* __restrict__ Wu,
             const float* __restrict__ Wo,
             float*       __restrict__ Zx)
{
    // [k-step][ntile][16B-half][lane] : 16*2*2*32*16B = 32 KB
    __shared__ v8h Bh[KSTEPS][2][2][32];

    const int lane    = threadIdx.x & 31;
    const int wave    = threadIdx.x >> 5;
    const int halfsel = lane >> 4;          // 0: lanes 0-15, 1: lanes 16-31
    const int lan     = lane & 15;

    // ---- Stage B (f32 -> f16) into LDS; wave w fills k-steps 2w, 2w+1.
    // B column n = lan (+16 for tile 1); lane half h holds K = 32*step+16h .. +15
    {
        const float* w0 = ((lan >> 3) == 0 ? Wf : Wi) + (size_t)(lan & 7) * DIN; // cols 0..15
        const float* w1 = ((lan >> 3) == 0 ? Wu : Wo) + (size_t)(lan & 7) * DIN; // cols 16..31
#pragma unroll
        for (int s = 0; s < 2; ++s) {
            const int step = wave * 2 + s;
            const int kb   = step * 32 + halfsel * 16;
            float4 b00 = *(const float4*)(w0 + kb);
            float4 b01 = *(const float4*)(w0 + kb + 4);
            float4 b02 = *(const float4*)(w0 + kb + 8);
            float4 b03 = *(const float4*)(w0 + kb + 12);
            Bh[step][0][0][lane] = cvt8(b00, b01);
            Bh[step][0][1][lane] = cvt8(b02, b03);
            float4 b10 = *(const float4*)(w1 + kb);
            float4 b11 = *(const float4*)(w1 + kb + 4);
            float4 b12 = *(const float4*)(w1 + kb + 8);
            float4 b13 = *(const float4*)(w1 + kb + 12);
            Bh[step][1][0][lane] = cvt8(b10, b11);
            Bh[step][1][1][lane] = cvt8(b12, b13);
        }
    }
    __syncthreads();

    // ---- A source: row (t*BATCH+b) = m0+lan; both lane halves share 16 token rows
    const int    m0   = (blockIdx.x * 8 + wave) * 16;
    const int    row  = m0 + lan;
    const int    tok  = sentence[row];
    const float* arow = emb + (size_t)tok * EMBED;

    v8f acc0 = {}; v8f acc1 = {};

    for (int step = 0; step < KSTEPS; ++step) {
        const int k0 = step * 32;
        // A fragment (16x32 f16): lane half h needs K = k0+16h..+15
        const float* pa = arow + k0 + halfsel * 16;
        float4 a0 = *(const float4*)(pa);
        float4 a1 = *(const float4*)(pa + 4);
        float4 a2 = *(const float4*)(pa + 8);
        float4 a3 = *(const float4*)(pa + 12);
        v16h a = cvt16(a0, a1, a2, a3);

        // B fragments straight from LDS (already f16, fragment layout)
        v16h b0 = join16(Bh[step][0][0][lane], Bh[step][0][1][lane]);
        v16h b1 = join16(Bh[step][1][0][lane], Bh[step][1][1][lane]);

        acc0 = __builtin_amdgcn_wmma_f32_16x16x32_f16(false, a, false, b0,
                                                      (short)0, acc0, false, false);
        acc1 = __builtin_amdgcn_wmma_f32_16x16x32_f16(false, a, false, b1,
                                                      (short)0, acc1, false, false);
    }

    // D layout: VGPR r, lane half h -> row m0 + r + 8h, col lan (+16 for tile 1)
    float* zb = Zx + (size_t)(m0 + halfsel * 8) * NCOLS + lan;
#pragma unroll
    for (int r = 0; r < 8; ++r) {
        zb[(size_t)r * NCOLS]      = acc0[r];
        zb[(size_t)r * NCOLS + 16] = acc1[r];
    }
}

__device__ __forceinline__ float sigmoidf_(float x) {
    return 1.0f / (1.0f + __expf(-x));
}

// Kernel 2: per-batch-row LSTM recurrence with the analytic qlayer:
//   c_j = cos(pre_j + th_j);  out_0 = c1..c7;  out_w = c0..cw (w>=1)
__global__ void __launch_bounds__(32)
qlstm_recur(const float* __restrict__ Zx,
            const float* __restrict__ Wf, const float* __restrict__ bf,
            const float* __restrict__ Wi, const float* __restrict__ bi,
            const float* __restrict__ Wu, const float* __restrict__ bu,
            const float* __restrict__ Wo, const float* __restrict__ bo,
            const float* __restrict__ thf, const float* __restrict__ thi,
            const float* __restrict__ thu, const float* __restrict__ tho,
            float* __restrict__ out)
{
    __shared__ float Wh[4][8][8];   // hidden-part weights W_g[:, 512:520]
    __shared__ float bias[4][8];
    __shared__ float th[4][8];

#pragma unroll
    for (int g = 0; g < 4; ++g) {
        const float* W  = (g == 0) ? Wf  : (g == 1) ? Wi  : (g == 2) ? Wu  : Wo;
        const float* bb = (g == 0) ? bf  : (g == 1) ? bi  : (g == 2) ? bu  : bo;
        const float* tt = (g == 0) ? thf : (g == 1) ? thi : (g == 2) ? thu : tho;
        if (threadIdx.x < 8) {
            int q = threadIdx.x;
#pragma unroll
            for (int j = 0; j < 8; ++j) Wh[g][q][j] = W[q * DIN + EMBED + j];
            bias[g][q] = bb[q];
            th[g][q]   = tt[q];
        }
    }
    __syncthreads();

    const int b = blockIdx.x * 32 + threadIdx.x;
    float hx[8], cx[8];
#pragma unroll
    for (int q = 0; q < 8; ++q) { hx[q] = 0.0f; cx[q] = 0.0f; }

    for (int t = 0; t < SEQ; ++t) {
        const float* z = Zx + ((size_t)t * BATCH + b) * NCOLS;
        float gates[4][8];
#pragma unroll
        for (int g = 0; g < 4; ++g) {
            float c[8];
#pragma unroll
            for (int q = 0; q < 8; ++q) {
                float pre = z[g * 8 + q] + bias[g][q];
#pragma unroll
                for (int j = 0; j < 8; ++j) pre = fmaf(Wh[g][q][j], hx[j], pre);
                c[q] = __cosf(pre + th[g][q]);
            }
            // qlayer: out0 = c1*...*c7 ; outw = c0*...*cw (w>=1)
            float outw[8];
            float p17 = c[1];
#pragma unroll
            for (int w = 2; w < 8; ++w) p17 *= c[w];
            outw[0] = p17;
            float run = c[0] * c[1];
            outw[1] = run;
#pragma unroll
            for (int w = 2; w < 8; ++w) { run *= c[w]; outw[w] = run; }
#pragma unroll
            for (int q = 0; q < 8; ++q)
                gates[g][q] = (g == 2) ? tanhf(outw[q]) : sigmoidf_(outw[q]);
        }
#pragma unroll
        for (int q = 0; q < 8; ++q) {
            cx[q] = gates[0][q] * cx[q] + gates[1][q] * gates[2][q];
            hx[q] = gates[3][q] * tanhf(cx[q]);
        }
    }

#pragma unroll
    for (int q = 0; q < 8; ++q) out[b * N_QUBITS + q] = hx[q];
}

extern "C" void kernel_launch(void* const* d_in, const int* in_sizes, int n_in,
                              void* d_out, int out_size, void* d_ws, size_t ws_size,
                              hipStream_t stream) {
    const int*   sentence = (const int*)  d_in[0];
    // d_in[1] = features: unused by the reference
    const float* emb = (const float*)d_in[2];
    const float* Wf  = (const float*)d_in[3];
    const float* bf  = (const float*)d_in[4];
    const float* Wi  = (const float*)d_in[5];
    const float* bi  = (const float*)d_in[6];
    const float* Wu  = (const float*)d_in[7];
    const float* bu  = (const float*)d_in[8];
    const float* Wo  = (const float*)d_in[9];
    const float* bo  = (const float*)d_in[10];
    const float* thf = (const float*)d_in[11];
    const float* thi = (const float*)d_in[12];
    const float* thu = (const float*)d_in[13];
    const float* tho = (const float*)d_in[14];

    float* Zx = (float*)d_ws;  // SEQ*BATCH*32 f32 = 4 MiB

    // 32768 M-rows / (8 waves * 16 rows) = 256 blocks
    qlstm_gemm_x<<<(SEQ * BATCH) / 128, 256, 0, stream>>>(
        sentence, emb, Wf, Wi, Wu, Wo, Zx);

    qlstm_recur<<<BATCH / 32, 32, 0, stream>>>(
        Zx, Wf, bf, Wi, bi, Wu, bu, Wo, bo, thf, thi, thu, tho, (float*)d_out);
}
